// HybridStateSpaceAttention_8598524526614
// MI455X (gfx1250) — matrix operand order repaired and checked by reading, not compiled, not run on hardware
//
#include <hip/hip_runtime.h>
#include <math.h>

// ---- problem constants ----
#define Bv 2
#define Sv 4096
#define Dmv 1024
#define Hh 8
#define HDv 128
#define WINv 512
#define STRIDEv 256
#define NWv 15
#define NGTv 64
#define SCALEv 0.088388347648318447f   // 1/sqrt(128)

typedef __attribute__((ext_vector_type(16))) __bf16 v16bf;
typedef __attribute__((ext_vector_type(8)))  float  v8f;

// ---- helpers ----
// native f32 -> bf16 (backend emits hardware cvt on gfx1250)
__device__ __forceinline__ __bf16 f2bf(float f) { return (__bf16)f; }

// contiguous 16 bf16 (32B) -> v16bf via two b128 loads (B-operand: col fixed, 16 consecutive K)
__device__ __forceinline__ v16bf ld16(const __bf16* p) {
  union { int4 q[2]; v16bf v; } u;
  u.q[0] = *(const int4*)p;
  u.q[1] = *(const int4*)(p + 8);
  return u.v;
}

// A-operand fragment: element e in [0,16) <-> K=e+16*hi
// => two contiguous 16B chunks at +16*hi and +16*hi+8. Works for global and LDS pointers.
__device__ __forceinline__ v16bf ld_afrag(const __bf16* p, int hi) {
  union { int4 q[2]; v16bf v; } u;
  const __bf16* b = p + (hi ? 16 : 0);
  u.q[0] = *(const int4*)b;
  u.q[1] = *(const int4*)(b + 8);
  return u.v;
}

// A-operand fragment from f32 source (pairs convert via v_cvt_pk_bf16_f32)
__device__ __forceinline__ v16bf ld_afrag_f32(const float* p, int hi) {
  const float* b = p + (hi ? 16 : 0);
  float4 a0 = *(const float4*)(b);
  float4 a1 = *(const float4*)(b + 4);
  float4 a2 = *(const float4*)(b + 8);
  float4 a3 = *(const float4*)(b + 12);
  v16bf v;
  v[0] = f2bf(a0.x);  v[1] = f2bf(a0.y);  v[2] = f2bf(a0.z);  v[3] = f2bf(a0.w);
  v[4] = f2bf(a1.x);  v[5] = f2bf(a1.y);  v[6] = f2bf(a1.z);  v[7] = f2bf(a1.w);
  v[8] = f2bf(a2.x);  v[9] = f2bf(a2.y);  v[10] = f2bf(a2.z); v[11] = f2bf(a2.w);
  v[12] = f2bf(a3.x); v[13] = f2bf(a3.y); v[14] = f2bf(a3.z); v[15] = f2bf(a3.w);
  return v;
}

__device__ __forceinline__ v8f wmma_bf16(v16bf a, v16bf b, v8f c) {
  return __builtin_amdgcn_wmma_f32_16x16x32_bf16(false, a, false, b, (short)0, c, false, false);
}

// reductions within a 16-lane half of a wave32 (rows live per-half in C/D layout)
__device__ __forceinline__ float hmax16(float v) {
  #pragma unroll
  for (int m = 1; m < 16; m <<= 1) v = fmaxf(v, __shfl_xor(v, m, 32));
  return v;
}
__device__ __forceinline__ float hsum16(float v) {
  #pragma unroll
  for (int m = 1; m < 16; m <<= 1) v += __shfl_xor(v, m, 32);
  return v;
}

// ============ K1: cast hidden -> bf16 shadow + (b,dm,s) transposed shadow; zero accumulator ============
__global__ void k_cast_zero(const float* __restrict__ x, __bf16* __restrict__ hbf,
                            __bf16* __restrict__ hT, float* __restrict__ loc) {
  int i0 = (blockIdx.x * 256 + threadIdx.x) * 4;
  #pragma unroll
  for (int j = 0; j < 4; ++j) {
    int i = i0 + j;
    __bf16 v = f2bf(x[i]);
    hbf[i] = v;
    loc[i] = 0.0f;
    int b = i >> 22;                       // / (S*Dm)
    int s = (i >> 10) & (Sv - 1);
    int d = i & (Dmv - 1);
    hT[((size_t)b * Dmv + d) * Sv + s] = v;
  }
}

// ============ K2: exponential diffusion, one (b, channel) column per WG in LDS ============
__global__ void k_bc(const float* __restrict__ x, float* __restrict__ bc) {
  __shared__ float cur[Sv];
  __shared__ float nxt[Sv];
  int b = blockIdx.x / Dmv;
  int d = blockIdx.x % Dmv;
  const float* src = x + (size_t)b * Sv * Dmv + d;
  float res[16];
  #pragma unroll
  for (int j = 0; j < 16; ++j) {
    int s = threadIdx.x + 256 * j;
    cur[s] = src[(size_t)s * Dmv];
    res[j] = 0.0f;
  }
  __syncthreads();
  float* a = cur; float* nb = nxt;
  for (int st = 1; st < Sv; st <<= 1) {
    #pragma unroll
    for (int j = 0; j < 16; ++j) {
      int s = threadIdx.x + 256 * j;
      float v = a[s] + 0.5f * (a[(s - st) & (Sv - 1)] + a[(s + st) & (Sv - 1)]);
      nb[s] = v;
      res[j] += v;
    }
    __syncthreads();
    float* t = a; a = nb; nb = t;
  }
  float* dst = bc + (size_t)b * Sv * Dmv + d;
  #pragma unroll
  for (int j = 0; j < 16; ++j)
    dst[(size_t)(threadIdx.x + 256 * j) * Dmv] = res[j] * (1.0f / 13.0f);
}

// ============ weight shadow casts (one-time, K-major per output column) ============
__global__ void k_wT_conv(const float* __restrict__ cw, __bf16* __restrict__ cwT) {
  int i = blockIdx.x * 256 + threadIdx.x;          // 1024*4096
  int o = i >> 12, kg = i & 4095;
  cwT[i] = f2bf(cw[(size_t)o * 4096 + (kg & 1023) * 4 + (kg >> 10)]);
}
__global__ void k_wT_mix(const float* __restrict__ mw, __bf16* __restrict__ mwT) {
  int i = blockIdx.x * 256 + threadIdx.x;          // 1024*2048 (n-major)
  int n = i >> 11, k = i & 2047;
  mwT[i] = f2bf(mw[(size_t)k * Dmv + n]);
}
__global__ void k_wT_out(const float* __restrict__ ow, __bf16* __restrict__ owT) {
  int i = blockIdx.x * 256 + threadIdx.x;          // 1024*1024 (n-major)
  int n = i >> 10, k = i & 1023;
  owT[i] = f2bf(ow[(size_t)k * Dmv + n]);
}

// ============ K3: conv1d(k=4,s=4) first 64 tokens as GEMM -> compressed KV (token-major + d-major) ============
__global__ void k_build_kv(const __bf16* __restrict__ hbf, const __bf16* __restrict__ cwT,
                           const float* __restrict__ cb, __bf16* __restrict__ kv,
                           __bf16* __restrict__ kvT) {
  int lane = threadIdx.x & 31, widx = threadIdx.x >> 5;
  int hi = lane >> 4, l15 = lane & 15;
  int wid = blockIdx.x * 4 + widx;          // 512 tiles: 8 mtiles x 64 ntiles
  int mtile = wid >> 6, ntile = wid & 63;
  int r = mtile * 16 + l15;
  int b = r >> 6, t = r & 63;
  const __bf16* arow = hbf + (size_t)(b * Sv + 4 * t) * Dmv;
  const __bf16* brow = cwT + (size_t)(ntile * 16 + l15) * 4096 + hi * 16;
  v8f acc = {};
  for (int kk = 0; kk < 4 * Dmv; kk += 32) {
    acc = wmma_bf16(ld_afrag(arow + kk, hi), ld16(brow + kk), acc);
  }
  #pragma unroll
  for (int e = 0; e < 8; ++e) {
    int m = e + hi * 8;
    int rr = mtile * 16 + m;
    int bb = rr >> 6, tt = rr & 63;
    int oo = ntile * 16 + l15;
    __bf16 c = f2bf(acc[e] + cb[oo]);
    int h = oo >> 7, dd = oo & 127;
    kv [(((size_t)bb * Hh + h) * 128 + tt) * HDv + dd] = c;
    kvT[(((size_t)bb * Hh + h) * HDv + dd) * 128 + tt] = c;
  }
}

// ============ K4: append broadcast global_memory tokens to KV (both layouts) ============
__global__ void k_copy_gmem(const float* __restrict__ gm, __bf16* __restrict__ kv,
                            __bf16* __restrict__ kvT) {
  int i = blockIdx.x * 256 + threadIdx.x;          // 2*8*64*128 = 131072
  int b = i >> 16;
  int rem = i & 65535;
  int h = rem >> 13, g = (rem >> 7) & 63, d = rem & 127;
  __bf16 v = f2bf(gm[((size_t)h * NGTv + g) * HDv + d]);
  kv [(((size_t)b * Hh + h) * 128 + NGTv + g) * HDv + d] = v;
  kvT[(((size_t)b * Hh + h) * HDv + d) * 128 + NGTv + g] = v;
}

// ============ K5: global attention: q(16xHD) vs 128 KV tokens, one q-tile per wave ============
__global__ void k_global_attn(const __bf16* __restrict__ hbf, const __bf16* __restrict__ kv,
                              const __bf16* __restrict__ kvT, float* __restrict__ glob) {
  __shared__ __align__(16) __bf16 pbuf[4 * 16 * 128];
  int lane = threadIdx.x & 31, widx = threadIdx.x >> 5;
  int hi = lane >> 4, l15 = lane & 15;
  int wid = blockIdx.x * 4 + widx;          // 4096 waves: (b,h) x 256 q-tiles
  int qt = wid & 255;
  int bh = wid >> 8;
  int b = bh >> 3, h = bh & 7;
  const __bf16* kvb  = kv  + ((size_t)b * Hh + h) * 128 * HDv;
  const __bf16* kvtb = kvT + ((size_t)b * Hh + h) * HDv * 128;

  const __bf16* qrow = hbf + (size_t)(b * Sv + qt * 16 + l15) * Dmv + h * HDv;
  v16bf qa[4];
  #pragma unroll
  for (int ks = 0; ks < 4; ++ks) qa[ks] = ld_afrag(qrow + ks * 32, hi);

  // S = Q K^T : 8 token tiles of 16
  v8f sc[8];
  #pragma unroll
  for (int tt = 0; tt < 8; ++tt) {
    v8f a = {};
    #pragma unroll
    for (int ks = 0; ks < 4; ++ks)
      a = wmma_bf16(qa[ks], ld16(kvb + (size_t)(tt * 16 + l15) * HDv + ks * 32 + hi * 16), a);
    sc[tt] = a;
  }
  // softmax over 128 tokens per row
  float rmax[8], rsum[8];
  #pragma unroll
  for (int e = 0; e < 8; ++e) {
    float mx = -3e38f;
    #pragma unroll
    for (int tt = 0; tt < 8; ++tt) mx = fmaxf(mx, sc[tt][e] * SCALEv);
    rmax[e] = hmax16(mx);
  }
  __bf16* pw = pbuf + widx * (16 * 128);
  #pragma unroll
  for (int e = 0; e < 8; ++e) {
    int m = e + hi * 8;
    float sum = 0.0f;
    #pragma unroll
    for (int tt = 0; tt < 8; ++tt) {
      float p = __expf(sc[tt][e] * SCALEv - rmax[e]);
      sum += p;
      pw[m * 128 + tt * 16 + l15] = f2bf(p);
    }
    rsum[e] = hsum16(sum);
  }
  __syncthreads();
  // O = P * KV   (B-operand from d-major kvT: 16 consecutive tokens per lane)
  #pragma unroll
  for (int dt = 0; dt < 8; ++dt) {
    v8f o = {};
    #pragma unroll
    for (int ks = 0; ks < 4; ++ks) {
      v16bf af = ld_afrag(pw + l15 * 128 + ks * 32, hi);
      v16bf bf = ld16(kvtb + (size_t)(dt * 16 + l15) * 128 + ks * 32 + hi * 16);
      o = wmma_bf16(af, bf, o);
    }
    #pragma unroll
    for (int e = 0; e < 8; ++e) {
      int m = e + hi * 8;
      glob[((size_t)b * Sv + qt * 16 + m) * Dmv + h * HDv + dt * 16 + l15] = o[e] / rsum[e];
    }
  }
}

// ============ K6: sliding-window attention, flash online softmax, tri-weighted atomic overlap-add ============
__global__ void k_sliding(const __bf16* __restrict__ hbf, const __bf16* __restrict__ hT,
                          float* __restrict__ loc) {
  __shared__ __align__(16) __bf16 pbuf[4 * 16 * 32];
  int lane = threadIdx.x & 31, widx = threadIdx.x >> 5;
  int hi = lane >> 4, l15 = lane & 15;
  int wid = blockIdx.x * 4 + widx;          // 7680 waves: b x 15 windows x 8 heads x 32 q-tiles
  int qt = wid & 31;
  int t1 = wid >> 5;
  int h = t1 & 7;
  int t2 = t1 >> 3;
  int n = t2 % 15;
  int b = t2 / 15;
  int base_s = n * STRIDEv;
  const __bf16* kb = hbf + (size_t)(b * Sv + base_s) * Dmv + h * HDv;  // K rows, stride Dm
  const __bf16* vT = hT + ((size_t)b * Dmv + h * HDv) * Sv + base_s;   // V cols, stride S

  const __bf16* qrow = hbf + (size_t)(b * Sv + base_s + qt * 16 + l15) * Dmv + h * HDv;
  v16bf qa[4];
  #pragma unroll
  for (int ks = 0; ks < 4; ++ks) qa[ks] = ld_afrag(qrow + ks * 32, hi);

  v8f oacc[8];
  #pragma unroll
  for (int dt = 0; dt < 8; ++dt) oacc[dt] = (v8f){};
  float mrun[8], lrun[8];
  #pragma unroll
  for (int e = 0; e < 8; ++e) { mrun[e] = -3e38f; lrun[e] = 0.0f; }
  __bf16* pw = pbuf + widx * (16 * 32);

  for (int c = 0; c < 16; ++c) {            // 16 chunks of 32 tokens
    int kb0 = c * 32;
    v8f s0 = {}, s1 = {};
    #pragma unroll
    for (int ks = 0; ks < 4; ++ks) {
      v16bf b0 = ld16(kb + (size_t)(kb0 + l15) * Dmv + ks * 32 + hi * 16);
      v16bf b1 = ld16(kb + (size_t)(kb0 + 16 + l15) * Dmv + ks * 32 + hi * 16);
      s0 = wmma_bf16(qa[ks], b0, s0);
      s1 = wmma_bf16(qa[ks], b1, s1);
    }
    float alpha[8];
    #pragma unroll
    for (int e = 0; e < 8; ++e) {
      float v0 = s0[e] * SCALEv, v1 = s1[e] * SCALEv;
      float nm = fmaxf(mrun[e], hmax16(fmaxf(v0, v1)));
      alpha[e] = __expf(mrun[e] - nm);
      mrun[e] = nm;
      float e0 = __expf(v0 - nm), e1 = __expf(v1 - nm);
      lrun[e] = lrun[e] * alpha[e] + hsum16(e0 + e1);
      int m = e + hi * 8;
      pw[m * 32 + l15]      = f2bf(e0);
      pw[m * 32 + 16 + l15] = f2bf(e1);
    }
    #pragma unroll
    for (int dt = 0; dt < 8; ++dt)
      #pragma unroll
      for (int e = 0; e < 8; ++e) oacc[dt][e] *= alpha[e];
    __syncthreads();
    v16bf af = ld_afrag(pw + l15 * 32, hi);
    #pragma unroll
    for (int dt = 0; dt < 8; ++dt) {
      v16bf bf = ld16(vT + (size_t)(dt * 16 + l15) * Sv + kb0 + hi * 16);
      oacc[dt] = wmma_bf16(af, bf, oacc[dt]);
    }
    __syncthreads();
  }
  #pragma unroll
  for (int dt = 0; dt < 8; ++dt)
    #pragma unroll
    for (int e = 0; e < 8; ++e) {
      int m = e + hi * 8;
      int s = base_s + qt * 16 + m;
      float tri = 0.5f + (float)(qt * 16 + m) * (1.0f / 511.0f);
      float v = oacc[dt][e] / lrun[e] * tri;
      atomicAdd(&loc[((size_t)b * Sv + s) * Dmv + h * HDv + dt * 16 + l15], v);
    }
}

// ============ K7: normalize overlap-add by analytic triangular denominator ============
__global__ void k_norm(float* __restrict__ loc) {
  int i0 = (blockIdx.x * 256 + threadIdx.x) * 4;
  #pragma unroll
  for (int j = 0; j < 4; ++j) {
    int i = i0 + j;
    int s = (i >> 10) & (Sv - 1);
    int n0 = s >> 8;
    float den = 0.0f;
    #pragma unroll
    for (int dn = 0; dn < 2; ++dn) {
      int n = n0 - dn;
      if (n >= 0 && n < NWv) {
        int jj = s - n * STRIDEv;
        if (jj < WINv) den += 0.5f + (float)jj * (1.0f / 511.0f);
      }
    }
    loc[i] /= den;
  }
}

// ============ K8: gate GEMM (K=2048) + sigmoid mix + bc add -> mixed (bf16) ============
__global__ void k_mix(const float* __restrict__ loc, const float* __restrict__ glob,
                      const float* __restrict__ bc, const __bf16* __restrict__ mwT,
                      const float* __restrict__ mb, __bf16* __restrict__ mixed) {
  int lane = threadIdx.x & 31, widx = threadIdx.x >> 5;
  int hi = lane >> 4, l15 = lane & 15;
  int wid = blockIdx.x * 4 + widx;          // 32768 tiles: 512 mtiles x 64 ntiles
  int mtile = wid >> 6, ntile = wid & 63;
  int r = mtile * 16 + l15;
  int nn = ntile * 16 + l15;
  const float* la = loc  + (size_t)r * Dmv;
  const float* gl = glob + (size_t)r * Dmv;
  const __bf16* brow = mwT + (size_t)nn * (2 * Dmv) + hi * 16;
  v8f acc = {};
  for (int kk = 0; kk < 2 * Dmv; kk += 32) {
    if (kk + 64 < 2 * Dmv)
      __builtin_prefetch(brow + kk + 64, 0, 1);
    const float* src = (kk < Dmv) ? (la + kk) : (gl + kk - Dmv);
    acc = wmma_bf16(ld_afrag_f32(src, hi), ld16(brow + kk), acc);
  }
  #pragma unroll
  for (int e = 0; e < 8; ++e) {
    size_t rr = (size_t)(mtile * 16 + e + hi * 8);
    float g = 1.0f / (1.0f + __expf(-(acc[e] + mb[nn])));
    float v = g * loc[rr * Dmv + nn] + (1.0f - g) * glob[rr * Dmv + nn] + bc[rr * Dmv + nn];
    mixed[rr * Dmv + nn] = f2bf(v);
  }
}

// ============ K9: output GEMM (K=1024) + bias -> d_out (f32) ============
__global__ void k_out(const __bf16* __restrict__ mixed, const __bf16* __restrict__ owT,
                      const float* __restrict__ ob, float* __restrict__ out) {
  int lane = threadIdx.x & 31, widx = threadIdx.x >> 5;
  int hi = lane >> 4, l15 = lane & 15;
  int wid = blockIdx.x * 4 + widx;
  int mtile = wid >> 6, ntile = wid & 63;
  int nn = ntile * 16 + l15;
  const __bf16* arow = mixed + (size_t)(mtile * 16 + l15) * Dmv;
  const __bf16* brow = owT + (size_t)nn * Dmv + hi * 16;
  v8f acc = {};
  for (int kk = 0; kk < Dmv; kk += 32) {
    if (kk + 64 < Dmv)
      __builtin_prefetch(brow + kk + 64, 0, 1);
    acc = wmma_bf16(ld_afrag(arow + kk, hi), ld16(brow + kk), acc);
  }
  #pragma unroll
  for (int e = 0; e < 8; ++e) {
    size_t rr = (size_t)(mtile * 16 + e + hi * 8);
    out[rr * Dmv + nn] = acc[e] + ob[nn];
  }
}

// ============ launcher ============
extern "C" void kernel_launch(void* const* d_in, const int* in_sizes, int n_in,
                              void* d_out, int out_size, void* d_ws, size_t ws_size,
                              hipStream_t stream) {
  (void)in_sizes; (void)n_in; (void)out_size; (void)ws_size;
  const float* x  = (const float*)d_in[0];
  const float* gm = (const float*)d_in[1];
  const float* cw = (const float*)d_in[2];
  const float* cb = (const float*)d_in[3];
  const float* mw = (const float*)d_in[4];
  const float* mb = (const float*)d_in[5];
  const float* ow = (const float*)d_in[6];
  const float* ob = (const float*)d_in[7];
  float* out = (float*)d_out;

  char* w = (char*)d_ws;
  __bf16* hbf = (__bf16*)w;                                  // 16 MiB (reused as mixed later)
  __bf16* hT  = (__bf16*)(w + (size_t)(16u  << 20));         // 16 MiB (b,dm,s) shadow
  float*  loc = (float*)(w + (size_t)(32u  << 20));          // 32 MiB
  float*  glb = (float*)(w + (size_t)(64u  << 20));          // 32 MiB
  float*  bc  = (float*)(w + (size_t)(96u  << 20));          // 32 MiB
  __bf16* kv  = (__bf16*)(w + (size_t)(128u << 20));         // 512 KiB (token-major)
  __bf16* kvT = (__bf16*)(w + (size_t)(128u << 20) + (512u << 10)); // 512 KiB (d-major)
  __bf16* cwT = (__bf16*)(w + (size_t)(129u << 20));         // 8 MiB
  __bf16* mwT = (__bf16*)(w + (size_t)(137u << 20));         // 4 MiB
  __bf16* owT = (__bf16*)(w + (size_t)(141u << 20));         // 2 MiB
  __bf16* mixed = hbf;                                       // safe reuse after K6

  k_cast_zero  <<<8192, 256, 0, stream>>>(x, hbf, hT, loc);
  k_bc         <<<Bv * Dmv, 256, 0, stream>>>(x, bc);
  k_wT_conv    <<<16384, 256, 0, stream>>>(cw, cwT);
  k_wT_mix     <<<8192, 256, 0, stream>>>(mw, mwT);
  k_wT_out     <<<4096, 256, 0, stream>>>(ow, owT);
  k_build_kv   <<<128, 128, 0, stream>>>(hbf, cwT, cb, kv, kvT);
  k_copy_gmem  <<<512, 256, 0, stream>>>(gm, kv, kvT);
  k_global_attn<<<1024, 128, 0, stream>>>(hbf, kv, kvT, glb);
  k_sliding    <<<1920, 128, 0, stream>>>(hbf, hT, loc);
  k_norm       <<<8192, 256, 0, stream>>>(loc);
  k_mix        <<<8192, 128, 0, stream>>>(loc, glb, bc, mwT, mb, mixed);
  k_out        <<<8192, 128, 0, stream>>>(mixed, owT, ob, out);
}
